// FlexibleLoss_86242943304060
// MI455X (gfx1250) — matrix operand order repaired and checked by reading, not compiled, not run on hardware
//
#include <hip/hip_runtime.h>
#include <math.h>

// ---------------------------------------------------------------------------
// Problem constants (match the reference)
// ---------------------------------------------------------------------------
#define BB   32
#define LL   512
#define DD   1024
#define NEGV (-1e7f)

// LDS panel geometry: 32 bf16 columns of W1^T, padded column stride.
// 2096 B = 16-byte aligned, and 2096/4 = 524 ≡ 12 (mod 64) -> the 16 column
// lanes of a b128 fragment read start on 16 distinct LDS banks.
#define COLB   2096
#define PANELB (32 * COLB)   // 67,072 B per panel

typedef __attribute__((ext_vector_type(16))) __bf16 v16bf;
typedef __attribute__((ext_vector_type(8)))  float  v8f;
typedef __attribute__((ext_vector_type(4)))  int    v4i;

#if __has_builtin(__builtin_amdgcn_global_load_async_to_lds_b128)
#define HAVE_ASYNC_LDS 1
#else
#define HAVE_ASYNC_LDS 0
#endif

#define GAS __attribute__((address_space(1)))
#define LAS __attribute__((address_space(3)))

__device__ __forceinline__ void async_copy16(const void* gsrc, void* ldst) {
#if HAVE_ASYNC_LDS
  __builtin_amdgcn_global_load_async_to_lds_b128(
      (GAS v4i*)gsrc, (LAS v4i*)ldst, 0, 0);
#else
  *(uint4*)ldst = *(const uint4*)gsrc;
#endif
}
__device__ __forceinline__ void wait_async_le16() {
#if HAVE_ASYNC_LDS
#if __has_builtin(__builtin_amdgcn_s_wait_asynccnt)
  __builtin_amdgcn_s_wait_asynccnt(16);
#else
  asm volatile("s_wait_asynccnt 0x10" ::: "memory");
#endif
#endif
}
__device__ __forceinline__ void wait_async_le0() {
#if HAVE_ASYNC_LDS
#if __has_builtin(__builtin_amdgcn_s_wait_asynccnt)
  __builtin_amdgcn_s_wait_asynccnt(0);
#else
  asm volatile("s_wait_asynccnt 0x0" ::: "memory");
#endif
#endif
}

// fp32 -> bf16 round-to-nearest-even
__device__ __forceinline__ unsigned short f2bf(float f) {
  unsigned int u = __float_as_uint(f);
  u = (u + 0x7FFFu + ((u >> 16) & 1u)) >> 16;
  return (unsigned short)u;
}
__device__ __forceinline__ unsigned int pack_bf16(float lo, float hi) {
  return (unsigned int)f2bf(lo) | ((unsigned int)f2bf(hi) << 16);
}

// ---------------------------------------------------------------------------
// Kernel 0: W1 (1024x1024 fp32, row-major [k][n]) -> W1^T (bf16, [n][k]).
// 64x64 tiles, LDS transpose, coalesced read & write.  One-time 2 MB output.
// ---------------------------------------------------------------------------
__global__ __launch_bounds__(256) void w1_transpose_bf16_kernel(
    const float* __restrict__ W1, unsigned int* __restrict__ W1Tp)
{
  __shared__ unsigned short sT[64][72];
  const int k0 = (blockIdx.x & 15) * 64;
  const int n0 = (blockIdx.x >> 4) * 64;
  for (int i = threadIdx.x; i < 4096; i += 256) {
    const int kr = i >> 6, nc = i & 63;
    sT[nc][kr] = f2bf(W1[(size_t)(k0 + kr) * DD + n0 + nc]);
  }
  __syncthreads();
  for (int i = threadIdx.x; i < 2048; i += 256) {
    const int nr = i >> 5, kp = i & 31;
    const unsigned int lo = sT[nr][kp * 2];
    const unsigned int hi = sT[nr][kp * 2 + 1];
    W1Tp[(size_t)(n0 + nr) * 512 + (k0 >> 1) + kp] = lo | (hi << 16);
  }
}

// ---------------------------------------------------------------------------
// Kernel 1: fused  relu(X*W1 + b1)*W2 + b2  ->  masked log-softmax.
// 256 threads (8 wave32s) per WG, 64 tokens per WG.
//   - X tile converted to bf16 in LDS once (128 KB)
//   - W1^T staged in double-buffered async LDS panels (2 x 65.5 KB)
//   - wave w: M-tile = w&3, N-group = w>>2; superstep i covers N-tiles {2i,2i+1}
// ---------------------------------------------------------------------------
__global__ __launch_bounds__(256) void ff_logprob_kernel(
    const float* __restrict__ X,  const int* __restrict__ seq_mask,
    const unsigned short* __restrict__ W1T,
    const float* __restrict__ b1,
    const float* __restrict__ W2, const float* __restrict__ b2,
    float* __restrict__ lp_out)
{
  __shared__ unsigned int sX[64 * 512];                    // 128 KB bf16 X tile
  __shared__ __align__(16) unsigned char sW[2][PANELB];    // 2 x 65.5 KB panels
  __shared__ float sLogits[64 * 3];

  const int tid  = threadIdx.x;
  const int tok0 = blockIdx.x * 64;

  // ---- stage X tile (fp32 -> bf16) into LDS ----
  for (int i = tid; i < 64 * 512; i += 256) {
    const int row = i >> 9;
    const int kp  = i & 511;
    const float2 x2 = *(const float2*)(X + (size_t)(tok0 + row) * DD + kp * 2);
    sX[row * 512 + kp] = pack_bf16(x2.x, x2.y);
  }
  if (tid < 192) sLogits[tid] = b2[tid % 3];

  // ---- issue async copy of panel 0 (16 b128 chunks per thread) ----
  const unsigned char* w1tb = (const unsigned char*)W1T;
  {
    const unsigned char* src = w1tb;                 // panel 0: cols 0..31
    #pragma unroll
    for (int j = 0; j < 16; ++j) {
      const int e  = j * 256 + tid;                  // 0..4095
      const int c  = e >> 7;                         // column 0..31
      const int kc = e & 127;                        // 16-byte chunk in column
      async_copy16(src + c * 2048 + kc * 16, &sW[0][c * COLB + kc * 16]);
    }
  }

  const int wave  = tid >> 5;
  const int lane  = tid & 31;
  const int mtile = wave & 3;
  const int ngrp  = wave >> 2;
  const int hi    = lane >> 4;
  const int col   = lane & 15;
  const int arow  = mtile * 16 + col;

  float acc[8][3];
  #pragma unroll
  for (int r = 0; r < 8; ++r)
    #pragma unroll
    for (int c = 0; c < 3; ++c) acc[r][c] = 0.f;

  const unsigned int* aBase = &sX[arow * 512];

  for (int i = 0; i < 32; ++i) {                     // 32 supersteps of 32 cols
    // prefetch next panel into the other buffer
    if (i + 1 < 32) {
      const unsigned char* src = w1tb + (size_t)(i + 1) * 32 * 2048;
      unsigned char* dst = sW[(i + 1) & 1];
      #pragma unroll
      for (int j = 0; j < 16; ++j) {
        const int e  = j * 256 + tid;
        const int c  = e >> 7;
        const int kc = e & 127;
        async_copy16(src + c * 2048 + kc * 16, dst + c * COLB + kc * 16);
      }
      wait_async_le16();       // our 16 new chunks may be in flight; panel i landed
    } else {
      wait_async_le0();
    }
    __syncthreads();

    const int nt = 2 * i + ngrp;                     // this wave's N-tile
    const int ng = nt * 16 + col;                    // lane's global column
    const unsigned char* bp = &sW[i & 1][(ngrp * 16 + col) * COLB];

    v8f cacc = {};
    #pragma unroll 4
    for (int k0 = 0; k0 < DD; k0 += 32) {
      union { v16bf v; uint4 q[2]; } A;
      A.q[0] = *(const uint4*)(aBase + (k0 >> 1) + hi * 4);
      A.q[1] = *(const uint4*)(aBase + (k0 >> 1) + 8 + hi * 4);
      union { v16bf v; uint4 q[2]; } Bf;
      const unsigned char* bk = bp + (size_t)(k0 + hi * 8) * 2;
      Bf.q[0] = *(const uint4*)(bk);
      Bf.q[1] = *(const uint4*)(bk + 32);
      cacc = __builtin_amdgcn_wmma_f32_16x16x32_bf16(
          false, A.v, false, Bf.v, (short)0, cacc, false, false);
    }

    // epilogue: h = relu(c + b1[n]); logits += h * W2[n][*]
    const float bn  = b1[ng];
    const float w20 = W2[ng * 3 + 0];
    const float w21 = W2[ng * 3 + 1];
    const float w22 = W2[ng * 3 + 2];
    #pragma unroll
    for (int r = 0; r < 8; ++r) {
      float h = cacc[r] + bn;
      h = h > 0.f ? h : 0.f;
      acc[r][0] += h * w20;
      acc[r][1] += h * w21;
      acc[r][2] += h * w22;
    }
    __syncthreads();                                 // protect buf before reuse
  }

  // reduce over the 16 column-lanes of each half-wave (wave32 shuffles)
  #pragma unroll
  for (int off = 8; off >= 1; off >>= 1) {
    #pragma unroll
    for (int r = 0; r < 8; ++r) {
      acc[r][0] += __shfl_xor(acc[r][0], off, 32);
      acc[r][1] += __shfl_xor(acc[r][1], off, 32);
      acc[r][2] += __shfl_xor(acc[r][2], off, 32);
    }
  }
  if (col == 0) {
    const int rbase = mtile * 16 + hi * 8;
    #pragma unroll
    for (int r = 0; r < 8; ++r) {
      atomicAdd(&sLogits[(rbase + r) * 3 + 0], acc[r][0]);
      atomicAdd(&sLogits[(rbase + r) * 3 + 1], acc[r][1]);
      atomicAdd(&sLogits[(rbase + r) * 3 + 2], acc[r][2]);
    }
  }
  __syncthreads();

  if (tid < 64) {
    const int tok = tok0 + tid;
    const float l0 = sLogits[tid * 3 + 0];
    const float l1 = sLogits[tid * 3 + 1];
    const float l2 = sLogits[tid * 3 + 2];
    const float m   = fmaxf(l0, fmaxf(l1, l2));
    const float lse = m + __logf(__expf(l0 - m) + __expf(l1 - m) + __expf(l2 - m));
    const float msk = (float)seq_mask[tok];
    lp_out[tok * 3 + 0] = (l0 - lse) * msk;
    lp_out[tok * 3 + 1] = (l1 - lse) * msk;
    lp_out[tok * 3 + 2] = (l2 - lse) * msk;
  }
}

// ---------------------------------------------------------------------------
// Kernel 2: marginal log-likelihood over BIO candidate sequences.
// ---------------------------------------------------------------------------
__global__ __launch_bounds__(256) void bio_loss_kernel(
    const float* __restrict__ lp,   const int* __restrict__ seq_mask,
    const int* __restrict__ ab,     const int* __restrict__ span,
    const int* __restrict__ is_bio, float* __restrict__ out)
{
  __shared__ float sSeq[9];
  __shared__ int   sRed[8];
  __shared__ int   sPre;
  const int b    = blockIdx.x;
  const int tid  = threadIdx.x;
  const int lane = tid & 31;
  const int wave = tid >> 5;

  int part = 0;
  for (int i = tid; i < 8 * LL; i += 256) {
    const int m = i >> 9, l = i & 511;
    part += ab[((b * 8) + m) * LL + l] * seq_mask[b * LL + l];
  }
  #pragma unroll
  for (int off = 16; off >= 1; off >>= 1) part += __shfl_xor(part, off, 32);
  if (lane == 0) sRed[wave] = part;
  __syncthreads();
  if (tid == 0) {
    int s = 0;
    for (int w = 0; w < 8; ++w) s += sRed[w];
    sPre = (s > 0) ? 1 : 0;
  }
  __syncthreads();
  const int pre = sPre;

  for (int row = wave; row < 9; row += 8) {
    float s = 0.f;
    int   ts = 0;
    for (int l = lane; l < LL; l += 32) {
      int tag;
      if (row < 8) tag = ab[((b * 8) + row) * LL + l] * seq_mask[b * LL + l];
      else         tag = span[b * LL + l] * (1 - pre);
      s  += lp[((size_t)(b * LL + l)) * 3 + tag];
      ts += tag;
    }
    #pragma unroll
    for (int off = 16; off >= 1; off >>= 1) {
      s  += __shfl_xor(s, off, 32);
      ts += __shfl_xor(ts, off, 32);
    }
    if (lane == 0) sSeq[row] = (ts > 0) ? s : NEGV;
  }
  __syncthreads();

  if (tid == 0) {
    float m = sSeq[0];
    for (int r = 1; r < 9; ++r) m = fmaxf(m, sSeq[r]);
    float sum = 0.f;
    for (int r = 0; r < 9; ++r) sum += __expf(sSeq[r] - m);
    out[b] = is_bio[b] ? (m + __logf(sum)) : NEGV;
  }
}

// ---------------------------------------------------------------------------
extern "C" void kernel_launch(void* const* d_in, const int* in_sizes, int n_in,
                              void* d_out, int out_size, void* d_ws, size_t ws_size,
                              hipStream_t stream) {
  (void)in_sizes; (void)n_in; (void)out_size; (void)ws_size;
  const float* X    = (const float*)d_in[0];   // bert_out (B,L,D)
  const int*   seq  = (const int*)  d_in[1];   // seq_mask (B,L)
  const int*   ab   = (const int*)  d_in[4];   // answer_as_list_of_bios (B,M,L)
  const int*   span = (const int*)  d_in[5];   // span_bio_labels (B,L)
  const int*   isb  = (const int*)  d_in[6];   // is_bio_mask (B,)
  const float* W1   = (const float*)d_in[7];
  const float* b1   = (const float*)d_in[8];
  const float* W2   = (const float*)d_in[9];
  const float* b2   = (const float*)d_in[10];

  // workspace layout: [0, 192KB) masked log-probs; [192KB, +2MB) W1^T bf16
  float*        lp   = (float*)d_ws;
  unsigned int* W1Tp = (unsigned int*)((char*)d_ws + 196608);
  float*        out  = (float*)d_out;

  w1_transpose_bf16_kernel<<<256, 256, 0, stream>>>(W1, W1Tp);
  ff_logprob_kernel<<<(BB * LL) / 64, 256, 0, stream>>>(
      X, seq, (const unsigned short*)W1Tp, b1, W2, b2, lp);
  bio_loss_kernel<<<BB, 256, 0, stream>>>(lp, seq, ab, span, isb, out);
}
